// GNN_auto_19086834664178
// MI455X (gfx1250) — matrix operand order repaired and checked by reading, not compiled, not run on hardware
//
#include <hip/hip_runtime.h>

#define DEVINL __device__ __forceinline__

typedef __attribute__((ext_vector_type(16))) _Float16 v16h;
typedef __attribute__((ext_vector_type(8)))  float    v8f;

namespace {
constexpr int N      = 64;
constexpr int N_NODE = 50000;
constexpr int E      = 250000;
constexpr int H      = 256;
constexpr int A      = 128;
constexpr int R      = 401;
constexpr int L      = 3;
constexpr int N_ENT  = 50000;
constexpr int K3     = 3 * H;   // 768: concat [hs|hr|hqr] K-dim
}

// ---------- WMMA fragment helpers (ISA 7.12.2 layouts, wave32) ----------
// A (16x32 f16): M = lane%16; half j=2v+p -> K = (v<4 ? 2v+p : 16+2(v-4)+p) + (lane>=16 ? 8 : 0)
DEVINL v16h load_a_frag(const _Float16* base, int stride_h, int m, int kb, bool hi) {
  union { v16h v; unsigned u[8]; } r;
  const unsigned* p = (const unsigned*)base;
  const unsigned row  = (unsigned)m * (unsigned)stride_h;
  const unsigned koff = hi ? 8u : 0u;
  const unsigned k0 = (row + (unsigned)kb + koff) >> 1;        // K = kb+koff+2v   (v=0..3)
  const unsigned k1 = (row + (unsigned)kb + 16u + koff) >> 1;  // K = kb+16+koff+2v
#pragma unroll
  for (int v = 0; v < 4; ++v) r.u[v] = p[k0 + v];
#pragma unroll
  for (int v = 0; v < 4; ++v) r.u[4 + v] = p[k1 + v];
  return r.v;
}
// B (32x16 f16), B[k][n] = W[n][k]: N = lane%16; half j -> K = kb + (j<8 ? j : 16+(j-8)) + (lane>=16 ? 8 : 0)
DEVINL v16h load_b_frag(const _Float16* base, int stride_h, int n_row, int kb, bool hi) {
  union { v16h v; unsigned u[8]; } r;
  const unsigned* p = (const unsigned*)base;
  const unsigned start = ((unsigned)n_row * (unsigned)stride_h + (unsigned)kb + (hi ? 8u : 0u)) >> 1;
#pragma unroll
  for (int v = 0; v < 4; ++v) r.u[v] = p[start + v];
#pragma unroll
  for (int v = 0; v < 4; ++v) r.u[4 + v] = p[start + 8 + v];
  return r.v;
}
DEVINL v8f wmma16(v16h a, v16h b, v8f c) {
  // v_wmma_f32_16x16x32_f16: 8 args (neg_a, A, neg_b, B, c_mod, C, reuse_a, reuse_b)
  return __builtin_amdgcn_wmma_f32_16x16x32_f16(false, a, false, b, (short)0, c, false, false);
}
DEVINL float sigmoidf_(float x) { return 1.f / (1.f + __expf(-x)); }

// ---------- Edge kernel: pre/alpha/message/scatter for 16 edges per block ----------
__global__ __launch_bounds__(256) void edge_kernel(
    const int* __restrict__ edges, const int* __restrict__ edge_batch,
    const int* __restrict__ q_rel, const float* __restrict__ hidden,
    const float* __restrict__ rela, const _Float16* __restrict__ W3,
    const float* __restrict__ wqr_b, const float* __restrict__ walpha_w,
    const float* __restrict__ walpha_b, float* __restrict__ agg)
{
  __shared__ _Float16 sA[16 * K3];   // 24 KB: [edge m][hs(256)|hr(256)|hqr(256)] f16
  __shared__ float s_alpha[16];
  __shared__ int   s_obj[16];
  const int tid  = threadIdx.x;
  const int lane = tid & 31, wave = tid >> 5;   // 8 waves
  const int e0 = blockIdx.x * 16;

  { // stage gathered rows as f16 (16 threads per edge)
    const int m = tid >> 4, st = tid & 15;
    const int e = e0 + m;
    const int sub = edges[e * 3 + 0];
    const int rel = edges[e * 3 + 1];
    if (st == 0) s_obj[m] = edges[e * 3 + 2];
    const int qr = q_rel[edge_batch[e]];
    const float* hs = hidden + (size_t)sub * H;
    const float* hr = rela + (size_t)rel * H;
    const float* hq = rela + (size_t)qr * H;
    _Float16* row = sA + m * K3;
#pragma unroll
    for (int c = st; c < H; c += 16) {
      row[c]         = (_Float16)hs[c];
      row[H + c]     = (_Float16)hr[c];
      row[2 * H + c] = (_Float16)hq[c];
    }
  }
  if (tid < 16) s_alpha[tid] = 0.f;
  __syncthreads();

  // pre tile: 16 edges x 16 alpha-cols per wave, K=768 -> 24 WMMAs
  const bool hi = lane >= 16;
  const int  hl = lane & 15;
  v8f acc = {};
#pragma unroll
  for (int kb = 0; kb < K3; kb += 32) {
    v16h a = load_a_frag(sA, K3, hl, kb, hi);
    v16h b = load_b_frag(W3, K3, wave * 16 + hl, kb, hi);
    acc = wmma16(a, b, acc);
  }
  // relu(pre)·walpha reduced over all 128 cols -> alpha per edge
  const int   acol = wave * 16 + hl;
  const float bias = wqr_b[acol];
  const float wa   = walpha_w[acol];
#pragma unroll
  for (int r = 0; r < 8; ++r) {
    float v = fmaxf(acc[r] + bias, 0.f) * wa;
    v += __shfl_xor(v, 1); v += __shfl_xor(v, 2);
    v += __shfl_xor(v, 4); v += __shfl_xor(v, 8);          // reduce within each 16-lane half
    if (hl == 0) atomicAdd(&s_alpha[r + (hi ? 8 : 0)], v); // combine 8 waves + both halves
  }
  __syncthreads();
  if (tid < 16) s_alpha[tid] = sigmoidf_(s_alpha[tid] + walpha_b[0]);
  __syncthreads();

  // message = alpha*hs*hr, segment-sum via f32 atomics into agg[obj]
  for (int idx = tid; idx < 16 * H; idx += 256) {
    const int m = idx >> 8, c = idx & (H - 1);
    const float hsv = (float)sA[m * K3 + c];
    const float hrv = (float)sA[m * K3 + H + c];
    atomicAdd(&agg[(size_t)s_obj[m] * H + c], s_alpha[m] * hsv * hrv);
  }
}

// ---------- fused node kernel: relu(agg@Wh^T) kept in LDS -> GRU -> hidden/h0 ----------
// 16-node tile per block, 16 waves (512 threads); wave w owns output cols [16w, 16w+16).
__global__ __launch_bounds__(512) void node_kernel(
    const float* __restrict__ agg, const _Float16* __restrict__ Wh16,
    const _Float16* __restrict__ Wih16, const _Float16* __restrict__ Whh16,
    const float* __restrict__ b_ih, const float* __restrict__ b_hh,
    float* __restrict__ hidden, float* __restrict__ h0)
{
  __shared__ _Float16 sA[16 * H];   // agg tile (f16)
  __shared__ _Float16 sN[16 * H];   // relu(agg@Wh^T) tile (f16), produced in-block
  __shared__ _Float16 sH[16 * H];   // h0 tile (f16)
  __shared__ int s_nz[16];
  const int tid = threadIdx.x;
  const int lane = tid & 31, wave = tid >> 5;
  const size_t base = (size_t)blockIdx.x * 16 * H;
  if (tid < 16) s_nz[tid] = 0;
  for (int idx = tid; idx < 16 * H; idx += 512) {
    sA[idx] = (_Float16)agg[base + idx];
    sH[idx] = (_Float16)h0[base + idx];
  }
  __syncthreads();

  const bool hi = lane >= 16;
  const int  hl = lane & 15;
  const int  col = wave * 16 + hl;

  // Phase 1: newhid tile = relu(agg @ Wh^T)  (8 WMMAs), exact nonzero flag from f32 acc
  v8f acc = {};
#pragma unroll
  for (int kb = 0; kb < H; kb += 32) {
    v16h a = load_a_frag(sA, H, hl, kb, hi);
    v16h b = load_b_frag(Wh16, H, col, kb, hi);
    acc = wmma16(a, b, acc);
  }
#pragma unroll
  for (int r = 0; r < 8; ++r) {
    const int m = r + (hi ? 8 : 0);
    const float v = fmaxf(acc[r], 0.f);
    sN[m * H + col] = (_Float16)v;
    if (v != 0.f) atomicOr(&s_nz[m], 1);   // relu output: row sum==0 <=> all zero
  }
  __syncthreads();

  // Phase 2: GRU gates, 6 accumulators x 8 K-steps = 48 WMMAs
  v8f ir = {}, iz = {}, in_ = {}, hr = {}, hz = {}, hn = {};
  for (int kb = 0; kb < H; kb += 32) {
    v16h aN = load_a_frag(sN, H, hl, kb, hi);
    v16h aH = load_a_frag(sH, H, hl, kb, hi);
    ir  = wmma16(aN, load_b_frag(Wih16, H, col,         kb, hi), ir);
    iz  = wmma16(aN, load_b_frag(Wih16, H, col + H,     kb, hi), iz);
    in_ = wmma16(aN, load_b_frag(Wih16, H, col + 2 * H, kb, hi), in_);
    hr  = wmma16(aH, load_b_frag(Whh16, H, col,         kb, hi), hr);
    hz  = wmma16(aH, load_b_frag(Whh16, H, col + H,     kb, hi), hz);
    hn  = wmma16(aH, load_b_frag(Whh16, H, col + 2 * H, kb, hi), hn);
  }
  const float bir = b_ih[col], biz = b_ih[col + H], bin = b_ih[col + 2 * H];
  const float bhr = b_hh[col], bhz = b_hh[col + H], bhn = b_hh[col + 2 * H];
#pragma unroll
  for (int r = 0; r < 8; ++r) {
    const int m = r + (hi ? 8 : 0);
    const size_t idx = base + (size_t)m * H + col;
    const float h0v  = h0[idx];
    const float rg   = sigmoidf_(ir[r] + bir + hr[r] + bhr);
    const float zg   = sigmoidf_(iz[r] + biz + hz[r] + bhz);
    const float nt   = tanhf(in_[r] + bin + rg * (hn[r] + bhn));
    const float hnew = (1.f - zg) * nt + zg * h0v;
    const float outv = s_nz[m] ? hnew : 0.f;
    hidden[idx] = outv;
    h0[idx]     = outv;
  }
}

// ---------- utility / pack / final ----------
__global__ void pack_half_kernel(const float* __restrict__ src, _Float16* __restrict__ dst, int n) {
  int i = blockIdx.x * blockDim.x + threadIdx.x;
  if (i < n) dst[i] = (_Float16)src[i];
}
__global__ void pack_w3_kernel(const float* __restrict__ Ws, const float* __restrict__ Wr,
                               const float* __restrict__ Wq, _Float16* __restrict__ W3) {
  int i = blockIdx.x * blockDim.x + threadIdx.x;  // over A*H
  if (i >= A * H) return;
  int a = i / H, k = i % H;
  W3[(size_t)a * K3 + k]         = (_Float16)Ws[i];
  W3[(size_t)a * K3 + H + k]     = (_Float16)Wr[i];
  W3[(size_t)a * K3 + 2 * H + k] = (_Float16)Wq[i];
}
__global__ void zero4_kernel(float4* p, int n4) {
  int i = blockIdx.x * blockDim.x + threadIdx.x;
  if (i < n4) p[i] = make_float4(0.f, 0.f, 0.f, 0.f);
}
__global__ void init_hidden_kernel(const int* __restrict__ q_rel, const int* __restrict__ qsub,
                                   const float* __restrict__ qrw, float* __restrict__ hidden) {
  hidden[(size_t)qsub[blockIdx.x] * H + threadIdx.x] =
      qrw[(size_t)q_rel[blockIdx.x] * H + threadIdx.x];
}
__global__ __launch_bounds__(256) void final_kernel(const float* __restrict__ hidden,
    const float* __restrict__ wf, const int* __restrict__ batch_idxs,
    const int* __restrict__ abs_idxs, float* __restrict__ out) {
  const int node = blockIdx.x * 8 + (threadIdx.x >> 5);
  const int lane = threadIdx.x & 31;
  const float* hrow = hidden + (size_t)node * H;
  float s = 0.f;
#pragma unroll
  for (int k = lane; k < H; k += 32) s += hrow[k] * wf[k];
  s += __shfl_xor(s, 16); s += __shfl_xor(s, 8); s += __shfl_xor(s, 4);
  s += __shfl_xor(s, 2);  s += __shfl_xor(s, 1);
  if (lane == 0) out[(size_t)batch_idxs[node] * N_ENT + abs_idxs[node]] = s;
}

extern "C" void kernel_launch(void* const* d_in, const int* in_sizes, int n_in,
                              void* d_out, int out_size, void* d_ws, size_t ws_size,
                              hipStream_t stream) {
  const int* q_rel        = (const int*)d_in[1];
  const int* batch_idxs   = (const int*)d_in[2];
  const int* abs_idxs     = (const int*)d_in[3];
  const int* qsub_idxs    = (const int*)d_in[4];
  const int* edge_batch   = (const int*)d_in[5];
  const int* edges        = (const int*)d_in[6];
  const float* rela_embed = (const float*)d_in[8];
  const float* Ws_w   = (const float*)d_in[9];
  const float* Wr_w   = (const float*)d_in[10];
  const float* Wqr_w  = (const float*)d_in[11];
  const float* Wqr_b  = (const float*)d_in[12];
  const float* walpha_w = (const float*)d_in[13];
  const float* walpha_b = (const float*)d_in[14];
  const float* Wh_w   = (const float*)d_in[15];
  const float* gru_w_ih = (const float*)d_in[16];
  const float* gru_w_hh = (const float*)d_in[17];
  const float* gru_b_ih = (const float*)d_in[18];
  const float* gru_b_hh = (const float*)d_in[19];
  const float* query_rela_w = (const float*)d_in[20];
  const float* W_final_w    = (const float*)d_in[21];
  float* out = (float*)d_out;
  (void)in_sizes; (void)n_in; (void)out_size; (void)ws_size;

  char* ws = (char*)d_ws;
  size_t off = 0;
  auto alloc = [&](size_t bytes) {
    void* p = ws + off;
    off += (bytes + 255) & ~(size_t)255;
    return p;
  };
  float*    hidden = (float*)alloc((size_t)N_NODE * H * 4);
  float*    h0     = (float*)alloc((size_t)N_NODE * H * 4);
  float*    agg    = (float*)alloc((size_t)N_NODE * H * 4);
  _Float16* W3_16  = (_Float16*)alloc((size_t)A * K3 * 2);
  _Float16* Wh16   = (_Float16*)alloc((size_t)H * H * 2);
  _Float16* Wih16  = (_Float16*)alloc((size_t)K3 * H * 2);
  _Float16* Whh16  = (_Float16*)alloc((size_t)K3 * H * 2);

  const int nodeF = N_NODE * H;   // 12.8M f32 per node buffer
  const int outF  = N * N_ENT;
  zero4_kernel<<<(nodeF / 4 + 255) / 256, 256, 0, stream>>>((float4*)hidden, nodeF / 4);
  zero4_kernel<<<(nodeF / 4 + 255) / 256, 256, 0, stream>>>((float4*)h0, nodeF / 4);
  zero4_kernel<<<(outF / 4 + 255) / 256, 256, 0, stream>>>((float4*)out, outF / 4);
  init_hidden_kernel<<<N, H, 0, stream>>>(q_rel, qsub_idxs, query_rela_w, hidden);
  pack_half_kernel<<<(K3 * H + 255) / 256, 256, 0, stream>>>(gru_w_ih, Wih16, K3 * H);
  pack_half_kernel<<<(K3 * H + 255) / 256, 256, 0, stream>>>(gru_w_hh, Whh16, K3 * H);

  for (int i = 0; i < L; ++i) {
    const float* rela = rela_embed + (size_t)i * R * H;
    pack_w3_kernel<<<(A * H + 255) / 256, 256, 0, stream>>>(
        Ws_w + (size_t)i * A * H, Wr_w + (size_t)i * A * H, Wqr_w + (size_t)i * A * H, W3_16);
    pack_half_kernel<<<(H * H + 255) / 256, 256, 0, stream>>>(Wh_w + (size_t)i * H * H, Wh16, H * H);
    zero4_kernel<<<(nodeF / 4 + 255) / 256, 256, 0, stream>>>((float4*)agg, nodeF / 4);
    edge_kernel<<<E / 16, 256, 0, stream>>>(edges, edge_batch, q_rel, hidden, rela, W3_16,
        Wqr_b + (size_t)i * A, walpha_w + (size_t)i * A, walpha_b + i, agg);
    node_kernel<<<N_NODE / 16, 512, 0, stream>>>(agg, Wh16, Wih16, Whh16,
        gru_b_ih, gru_b_hh, hidden, h0);
  }
  final_kernel<<<N_NODE / 8, 256, 0, stream>>>(hidden, W_final_w, batch_idxs, abs_idxs, out);
}
